// MAPOCA_45552423141838
// MI455X (gfx1250) — matrix operand order repaired and checked by reading, not compiled, not run on hardware
//
#include <hip/hip_runtime.h>
#include <math.h>

// ---------------------------------------------------------------------------
// MAPOCA forward for MI455X (gfx1250, wave32, WMMA 16x16x32 f16 -> f32 acc)
// B=48, N=16, OBS=128, ACT=32, HID=QHID=128, HEADS=4, HD=32, L=768, ROWS=12288
// All GEMM operands staged as f16 (weights pre-converted once); f32 accumulate.
// Fragments loaded as 128-bit vectors (global b128 / ds b128).
// ---------------------------------------------------------------------------

typedef __attribute__((ext_vector_type(16))) _Float16 v16h;
typedef __attribute__((ext_vector_type(8)))  _Float16 h8;
typedef __attribute__((ext_vector_type(8)))  float    v8f;

#define Bv   48
#define Nv   16
#define OBS  128
#define ACT  32
#define HID  128
#define QH   128
#define HD   32
#define Lv   (Bv * Nv)          // 768
#define ROWS (Lv * Nv)          // 12288

// f16 weight arena offsets (elements, all multiples of 8 -> 16B aligned)
#define OFF_AW1 0               // 256x128
#define OFF_AW2 32768           // 128x256
#define OFF_AW3 65536           // 32x128
#define OFF_EOW 69632           // 128x128
#define OFF_EAW 86016           // 128x160
#define OFF_RIW 106496          // 128x128
#define OFF_MIW 122880          // 384x128
#define OFF_MOW 172032          // 128x128
#define OFF_ROW 188416          // 128x128
#define W16_TOTAL 204800

__device__ __forceinline__ v16h cat16h(h8 lo, h8 hi) {
  return __builtin_shufflevector(lo, hi, 0, 1, 2, 3, 4, 5, 6, 7,
                                         8, 9, 10, 11, 12, 13, 14, 15);
}

// A fragment (16x32 f16): lane row = ln&15; K run = kA..kA+15 contiguous
__device__ __forceinline__ v16h load_a16(const _Float16* base, int stride) {
  const int ln = threadIdx.x & 31;
  const _Float16* p = base + (ln & 15) * stride + ((ln >> 4) * 16);
  return cat16h(*(const h8*)p, *(const h8*)(p + 8));
}

// B fragment (32x16 f16): lane col = ln&15; K run = kB..kB+15 contiguous
__device__ __forceinline__ v16h load_b16(const _Float16* base, int stride) {
  const int ln = threadIdx.x & 31;
  const _Float16* p = base + (ln & 15) * stride + ((ln >> 4) * 16);
  return cat16h(*(const h8*)p, *(const h8*)(p + 8));
}

__device__ __forceinline__ v8f wmma16(v16h a, v16h b, v8f acc) {
  return __builtin_amdgcn_wmma_f32_16x16x32_f16(false, a, false, b,
                                                (short)0, acc, false, false);
}

// C = X(16xK) @ W^T tile; X row-major f16 (stride xs), W f16 row-major (out,K)
__device__ __forceinline__ v8f wmma_tile16(const _Float16* X, int xs,
                                           const _Float16* W, int wstride, int K) {
  v8f acc = {};
  for (int k0 = 0; k0 < K; k0 += 32)
    acc = wmma16(load_a16(X + k0, xs), load_b16(W + k0, wstride), acc);
  return acc;
}

// C-fragment scatter: st(row, ncol, value); row = v + 8*half, col = lane&15
template <typename StF>
__device__ __forceinline__ void store_tile(const v8f& c, StF&& st) {
  const int ln = threadIdx.x & 31;
  const int half = ln >> 4;
  const int col = ln & 15;
#pragma unroll
  for (int v = 0; v < 8; ++v) st(v + 8 * half, col, c[v]);
}

__device__ __forceinline__ float leaky(float x) { return x > 0.f ? x : 0.01f * x; }
__device__ __forceinline__ float gelu_exact(float x) {
  return 0.5f * x * (1.0f + erff(x * 0.7071067811865476f));
}

// ---------------------------------------------------------------------------
// Kernel 0: convert all weight matrices f32 -> f16 arena.
// ---------------------------------------------------------------------------
__global__ void __launch_bounds__(256)
k_cvt_weights(const float* __restrict__ aw1, const float* __restrict__ aw2,
              const float* __restrict__ aw3, const float* __restrict__ eow,
              const float* __restrict__ eaw, const float* __restrict__ riw,
              const float* __restrict__ miw, const float* __restrict__ mow,
              const float* __restrict__ roww, _Float16* __restrict__ w16) {
  const int i = blockIdx.x * 256 + threadIdx.x;
  const float* src;
  int off;
  if      (i < OFF_AW2)   { src = aw1;  off = OFF_AW1; }
  else if (i < OFF_AW3)   { src = aw2;  off = OFF_AW2; }
  else if (i < OFF_EOW)   { src = aw3;  off = OFF_AW3; }
  else if (i < OFF_EAW)   { src = eow;  off = OFF_EOW; }
  else if (i < OFF_RIW)   { src = eaw;  off = OFF_EAW; }
  else if (i < OFF_MIW)   { src = riw;  off = OFF_RIW; }
  else if (i < OFF_MOW)   { src = miw;  off = OFF_MIW; }
  else if (i < OFF_ROW)   { src = mow;  off = OFF_MOW; }
  else if (i < W16_TOTAL) { src = roww; off = OFF_ROW; }
  else return;
  w16[i] = (_Float16)src[i - off];
}

// ---------------------------------------------------------------------------
// Kernel 1: actor MLP + state encoders. One wave per 16-row tile of (768,*).
// ---------------------------------------------------------------------------
__global__ void __launch_bounds__(32)
k_actor_encoder(const float* __restrict__ obs, const _Float16* __restrict__ w16,
                const float* __restrict__ ab1, const float* __restrict__ ab2,
                const float* __restrict__ ab3, const float* __restrict__ eob,
                const float* __restrict__ eab,
                float* __restrict__ out_policy,       // (768,32) f32
                float* __restrict__ out_eo,           // (768,128) f32
                _Float16* __restrict__ eo16,          // (768,128) f16
                _Float16* __restrict__ ea16)          // (768,128) f16
{
  __shared__ _Float16 cat[16][OBS + ACT];   // obs | policy (160)
  __shared__ _Float16 h1[16][2 * HID];      // 256
  __shared__ _Float16 h2[16][HID];          // 128
  const int r0 = blockIdx.x * 16;
  const int ln = threadIdx.x;

  // stage obs tile -> cat[:, 0:128] (f16)
  for (int idx = ln; idx < 16 * OBS; idx += 32) {
    const int r = idx >> 7, c = idx & 127;
    cat[r][c] = (_Float16)obs[(r0 + r) * OBS + c];
  }
  __syncthreads();

  // hidden_layer1: 128->256, leaky_relu
  for (int ct = 0; ct < 16; ++ct) {
    const int co = ct * 16;
    v8f acc = wmma_tile16(&cat[0][0], OBS + ACT, w16 + OFF_AW1 + co * OBS, OBS, OBS);
    store_tile(acc, [&](int m, int n, float x) {
      h1[m][co + n] = (_Float16)leaky(x + ab1[co + n]);
    });
  }
  __syncthreads();

  // hidden_layer2: 256->128, leaky_relu
  for (int ct = 0; ct < 8; ++ct) {
    const int co = ct * 16;
    v8f acc = wmma_tile16(&h1[0][0], 2 * HID, w16 + OFF_AW2 + co * 2 * HID,
                          2 * HID, 2 * HID);
    store_tile(acc, [&](int m, int n, float x) {
      h2[m][co + n] = (_Float16)leaky(x + ab2[co + n]);
    });
  }
  __syncthreads();

  // action_layer: 128->32, exact GELU -> policy (f32 out + f16 into cat[:,128:])
  for (int ct = 0; ct < 2; ++ct) {
    const int co = ct * 16;
    v8f acc = wmma_tile16(&h2[0][0], HID, w16 + OFF_AW3 + co * HID, HID, HID);
    store_tile(acc, [&](int m, int n, float x) {
      const float g = gelu_exact(x + ab3[co + n]);
      out_policy[(r0 + m) * ACT + co + n] = g;
      cat[m][OBS + co + n] = (_Float16)g;
    });
  }
  __syncthreads();

  // obs_encode_layer: 128->128 -> eo (f32 output + f16 copy)
  for (int ct = 0; ct < 8; ++ct) {
    const int co = ct * 16;
    v8f acc = wmma_tile16(&cat[0][0], OBS + ACT, w16 + OFF_EOW + co * HID, HID, OBS);
    store_tile(acc, [&](int m, int n, float x) {
      const float y = x + eob[co + n];
      out_eo[(r0 + m) * HID + co + n] = y;
      eo16[(r0 + m) * HID + co + n] = (_Float16)y;
    });
  }

  // obs_act_encode_layer: concat(obs,policy) 160 -> 128 -> ea (f16)
  for (int ct = 0; ct < 8; ++ct) {
    const int co = ct * 16;
    v8f acc = wmma_tile16(&cat[0][0], OBS + ACT, w16 + OFF_EAW + co * (OBS + ACT),
                          OBS + ACT, OBS + ACT);
    store_tile(acc, [&](int m, int n, float x) {
      ea16[(r0 + m) * HID + co + n] = (_Float16)(x + eab[co + n]);
    });
  }
}

// ---------------------------------------------------------------------------
// Kernel 2: counterfactual tile -> x1 = relu(. @ riw^T) -> in_proj (q|k|vT).
// One wave per l = b*16+i. V is written transposed: (n,h,d,key)-major.
// ---------------------------------------------------------------------------
__global__ void __launch_bounds__(32)
k_rsa_input(const _Float16* __restrict__ eo16, const _Float16* __restrict__ ea16,
            const _Float16* __restrict__ w16,
            const float* __restrict__ rib, const float* __restrict__ mib,
            float* __restrict__ x1buf,        // (12288,128) f32
            _Float16* __restrict__ qk16,      // (12288,256) f16  q|k
            _Float16* __restrict__ vT16)      // (16*4*32, 768) f16
{
  __shared__ _Float16 cft[16][HID];
  __shared__ _Float16 x1t[16][HID];
  const int l = blockIdx.x;               // 0..767
  const int b = l >> 4, i = l & 15;
  const int ln = threadIdx.x;

  // stage counterfactual rows: j==0 -> eo[b,i]; j>=1 -> ea[b, (j-1<i ? j-1 : j)]
  for (int idx = ln; idx < 16 * 16; idx += 32) {   // 16 rows x 16 chunks of 8
    const int j = idx >> 4, ch = idx & 15;
    const _Float16* src;
    if (j == 0) src = eo16 + (b * Nv + i) * HID;
    else {
      const int jm = j - 1;
      const int m = (jm < i) ? jm : j;
      src = ea16 + (b * Nv + m) * HID;
    }
    *(h8*)&cft[j][ch * 8] = *(const h8*)(src + ch * 8);
  }
  __syncthreads();

  // x1 = relu(cf @ riw^T + rib)
  for (int ct = 0; ct < 8; ++ct) {
    const int co = ct * 16;
    v8f acc = wmma_tile16(&cft[0][0], HID, w16 + OFF_RIW + co * HID, HID, HID);
    store_tile(acc, [&](int m, int n, float x) {
      const float y = fmaxf(x + rib[co + n], 0.f);
      x1buf[(l * 16 + m) * HID + co + n] = y;
      x1t[m][co + n] = (_Float16)y;
    });
  }
  __syncthreads();

  // in_proj: 128 -> 384; q,k kept row-major, v transposed to (n,h,d,key)
  for (int ct = 0; ct < 24; ++ct) {
    const int co = ct * 16;
    v8f acc = wmma_tile16(&x1t[0][0], HID, w16 + OFF_MIW + co * QH, QH, QH);
    store_tile(acc, [&](int m, int n, float x) {
      const int f = co + n;
      const float y = x + mib[f];
      if (f < 2 * QH) {
        qk16[(size_t)(l * 16 + m) * (2 * QH) + f] = (_Float16)y;
      } else {
        const int ff = f - 2 * QH;               // h*32+d
        vT16[((size_t)m * QH + ff) * Lv + l] = (_Float16)y;   // m == agent n
      }
    });
  }
}

// ---------------------------------------------------------------------------
// Kernel 3: flash attention over seq L=768 per (n,h). grid = (48 qtiles, 64).
// Per 32-key step: 2 WMMA (S = Q K^T) + online softmax + 2 WMMA (P V).
// ---------------------------------------------------------------------------
__global__ void __launch_bounds__(32)
k_attention(const _Float16* __restrict__ qk16,  // (12288,256)
            const _Float16* __restrict__ vT16,  // (2048,768)
            _Float16* __restrict__ ao16)        // (12288,128)
{
  __shared__ _Float16 pls[16][32];
  const int qt = blockIdx.x;                    // 0..47
  const int nh = blockIdx.y;                    // 0..63
  const int n = nh >> 2, h = nh & 3;

  const int ln = threadIdx.x & 31;
  const int half = ln >> 4;
  const int col = ln & 15;
  const float scale = 0.1767766952966369f;      // 1/sqrt(32)

  // Q fragment: rows = 16 queries (row stride 16*256), K-dim = head dim 32
  const v16h qa =
      load_a16(qk16 + ((size_t)(qt * 16) * Nv + n) * (2 * QH) + h * HD, Nv * 2 * QH);

  v8f o0 = {}, o1 = {};
  float mrow[8], lrow[8];
#pragma unroll
  for (int v = 0; v < 8; ++v) { mrow[v] = -1e30f; lrow[v] = 0.f; }

  for (int kt = 0; kt < Lv / 32; ++kt) {        // 24 steps of 32 keys
    const int k0 = kt * 32;

    // S = Q K^T : two 16-key fragments (B cols = keys, K run = head dim)
    v8f s0, s1;
#pragma unroll
    for (int kh = 0; kh < 2; ++kh) {
      const _Float16* kbase =
          qk16 + ((size_t)(k0 + kh * 16) * Nv + n) * (2 * QH) + QH + h * HD;
      const v16h bk = load_b16(kbase, Nv * 2 * QH);
      v8f z = {};
      const v8f r = wmma16(qa, bk, z);
      if (kh == 0) s0 = r; else s1 = r;
    }

    // online softmax (row-wise; 16-lane butterflies keep halves separate)
    float p0[8], p1[8];
#pragma unroll
    for (int v = 0; v < 8; ++v) {
      s0[v] *= scale; s1[v] *= scale;
      float mt = fmaxf(s0[v], s1[v]);
#pragma unroll
      for (int msk = 1; msk < 16; msk <<= 1) mt = fmaxf(mt, __shfl_xor(mt, msk, 32));
      const float mn = fmaxf(mrow[v], mt);
      const float corr = expf(mrow[v] - mn);
      mrow[v] = mn;
      lrow[v] *= corr;
      o0[v] *= corr; o1[v] *= corr;
      p0[v] = expf(s0[v] - mn);
      p1[v] = expf(s1[v] - mn);
      float rs = p0[v] + p1[v];
#pragma unroll
      for (int msk = 1; msk < 16; msk <<= 1) rs += __shfl_xor(rs, msk, 32);
      lrow[v] += rs;
    }

    // P (C layout) -> LDS -> A fragment
    __syncthreads();
#pragma unroll
    for (int v = 0; v < 8; ++v) {
      pls[v + 8 * half][col]      = (_Float16)p0[v];
      pls[v + 8 * half][16 + col] = (_Float16)p1[v];
    }
    __syncthreads();
    const v16h pa = load_a16(&pls[0][0], 32);

    // O += P V : B cols = d (stride Lv), K run = keys (contiguous in vT)
#pragma unroll
    for (int dh = 0; dh < 2; ++dh) {
      const _Float16* vbase =
          vT16 + ((size_t)(n * 4 + h) * HD + dh * 16) * Lv + k0;
      const v16h bv = load_b16(vbase, Lv);
      if (dh == 0) o0 = wmma16(pa, bv, o0);
      else         o1 = wmma16(pa, bv, o1);
    }
  }

  // normalize + store ao[lq, n, h*32+d] (f16)
#pragma unroll
  for (int v = 0; v < 8; ++v) {
    const float inv = 1.f / lrow[v];
    const int lq = qt * 16 + v + 8 * half;
    ao16[((size_t)lq * Nv + n) * QH + h * HD + col]      = (_Float16)(o0[v] * inv);
    ao16[((size_t)lq * Nv + n) * QH + h * HD + 16 + col] = (_Float16)(o1[v] * inv);
  }
}

// ---------------------------------------------------------------------------
// Kernel 4: out_proj + residual -> output layer (relu) -> q_layer -> sum_j.
// ---------------------------------------------------------------------------
__global__ void __launch_bounds__(32)
k_finalize(const _Float16* __restrict__ ao16, const float* __restrict__ x1buf,
           const _Float16* __restrict__ w16,
           const float* __restrict__ mob, const float* __restrict__ rob,
           const float* __restrict__ qw, const float* __restrict__ qb,
           float* __restrict__ out_q)             // (768)
{
  __shared__ _Float16 tls[16][QH];
  __shared__ _Float16 cls[16][QH];
  const int l = blockIdx.x;                      // 0..767

  // t = ao @ mow^T + mob + x1
  for (int ct = 0; ct < 8; ++ct) {
    const int co = ct * 16;
    v8f acc = wmma_tile16(ao16 + (size_t)(l * 16) * QH, QH,
                          w16 + OFF_MOW + co * QH, QH, QH);
    store_tile(acc, [&](int m, int n, float x) {
      tls[m][co + n] =
          (_Float16)(x + mob[co + n] + x1buf[(l * 16 + m) * QH + co + n]);
    });
  }
  __syncthreads();

  // ctx = relu(t @ row^T + rob)
  for (int ct = 0; ct < 8; ++ct) {
    const int co = ct * 16;
    v8f acc = wmma_tile16(&tls[0][0], QH, w16 + OFF_ROW + co * QH, QH, QH);
    store_tile(acc, [&](int m, int n, float x) {
      cls[m][co + n] = (_Float16)fmaxf(x + rob[co + n], 0.f);
    });
  }
  __syncthreads();

  // q_values[l] = sum_j (ctx[j] . qw + qb)
  const int ln = threadIdx.x & 31;
  float acc = 0.f;
  if (ln < 16) {
    for (int k = 0; k < QH; ++k) acc += (float)cls[ln][k] * qw[k];
  }
#pragma unroll
  for (int msk = 1; msk < 16; msk <<= 1) acc += __shfl_xor(acc, msk, 32);
  if (ln == 0) out_q[l] = acc + 16.0f * qb[0];
}

// ---------------------------------------------------------------------------
extern "C" void kernel_launch(void* const* d_in, const int* in_sizes, int n_in,
                              void* d_out, int out_size, void* d_ws, size_t ws_size,
                              hipStream_t stream) {
  const float* obs = (const float*)d_in[0];
  const float* aw1 = (const float*)d_in[1];  const float* ab1 = (const float*)d_in[2];
  const float* aw2 = (const float*)d_in[3];  const float* ab2 = (const float*)d_in[4];
  const float* aw3 = (const float*)d_in[5];  const float* ab3 = (const float*)d_in[6];
  const float* eow = (const float*)d_in[7];  const float* eob = (const float*)d_in[8];
  const float* eaw = (const float*)d_in[9];  const float* eab = (const float*)d_in[10];
  const float* riw = (const float*)d_in[11]; const float* rib = (const float*)d_in[12];
  const float* roww= (const float*)d_in[13]; const float* rob = (const float*)d_in[14];
  const float* miw = (const float*)d_in[15]; const float* mib = (const float*)d_in[16];
  const float* mow = (const float*)d_in[17]; const float* mob = (const float*)d_in[18];
  const float* qw  = (const float*)d_in[19]; const float* qb  = (const float*)d_in[20];

  float* out = (float*)d_out;
  float* out_policy = out;                 // 768*32
  float* out_q      = out + 24576;         // 768
  float* out_eo     = out + 25344;         // 768*128

  // workspace: f32 region first, then 16B-aligned f16 arena
  float* x1buf = (float*)d_ws;                                  // 12288*128 f32
  _Float16* wsh = (_Float16*)(x1buf + (size_t)ROWS * HID);
  _Float16* w16  = wsh;                                         // 204800
  _Float16* eo16 = w16  + W16_TOTAL;                            // 98304
  _Float16* ea16 = eo16 + (size_t)Lv * HID;                     // 98304
  _Float16* qk16 = ea16 + (size_t)Lv * HID;                     // 12288*256
  _Float16* vT16 = qk16 + (size_t)ROWS * 2 * QH;                // 2048*768
  _Float16* ao16 = vT16 + (size_t)16 * QH * Lv;                 // 12288*128

  k_cvt_weights<<<(W16_TOTAL + 255) / 256, 256, 0, stream>>>(
      aw1, aw2, aw3, eow, eaw, riw, miw, mow, roww, w16);
  k_actor_encoder<<<48, 32, 0, stream>>>(obs, w16, ab1, ab2, ab3, eob, eab,
                                         out_policy, out_eo, eo16, ea16);
  k_rsa_input<<<Lv, 32, 0, stream>>>(eo16, ea16, w16, rib, mib,
                                     x1buf, qk16, vT16);
  k_attention<<<dim3(48, 64), 32, 0, stream>>>(qk16, vT16, ao16);
  k_finalize<<<Lv, 32, 0, stream>>>(ao16, x1buf, w16, mob, rob, qw, qb, out_q);
}
